// NHGrid_34196529611041
// MI455X (gfx1250) — compile-verified, matrix operand order checked by reading, not run on hardware
//
#include <hip/hip_runtime.h>

typedef __attribute__((ext_vector_type(16))) _Float16 v16h;
typedef __attribute__((ext_vector_type(8)))  float    v8f;
typedef __attribute__((ext_vector_type(2)))  _Float16 h2;

#define H_MASK ((1 << 19) - 1)   // hashSize 524288

// WMMA fragment K index for element-pair i (0..7), hi = lane>>4 (ISA 7.12.2, 16-bit A 16x32)
__device__ __forceinline__ int frag_k(int i, int hi) {
    return (i < 4) ? (hi * 8 + 2 * i) : (16 + hi * 8 + 2 * (i - 4));
}

// A fragment: 16x32 f16, rows = points (row-major LDS activation buffer)
__device__ __forceinline__ v16h load_a(const _Float16* src, int stride, int row0, int k0, int lane) {
    int m = lane & 15, hi = lane >> 4;
    const h2* p = (const h2*)(src + (row0 + m) * stride + k0);
    v16h a;
#pragma unroll
    for (int i = 0; i < 8; ++i) {
        h2 v = p[frag_k(i, hi) >> 1];
        a[2 * i] = v[0]; a[2 * i + 1] = v[1];
    }
    return a;
}

// B fragment: 32x16 f16 from LDS weights stored transposed [Nout][Kpad] (K contiguous)
__device__ __forceinline__ v16h load_b(const _Float16* Wt, int Kpad, int Nout, int c0, int k0, int lane) {
    int n = c0 + (lane & 15), hi = lane >> 4;
    v16h b = {};
    if (n < Nout) {
        const h2* p = (const h2*)(Wt + n * Kpad + k0);
#pragma unroll
        for (int i = 0; i < 8; ++i) {
            h2 v = p[frag_k(i, hi) >> 1];
            b[2 * i] = v[0]; b[2 * i + 1] = v[1];
        }
    }
    return b;
}

// D/C: lane = N column, VGPR j = row (j + 8*(lane>>4)). Add bias, leaky-relu, store f16.
__device__ __forceinline__ void store_act(v8f acc, _Float16* dst, int stride, int row0, int c0,
                                          const float* bias, int Nout, int lane) {
    int n  = c0 + (lane & 15);
    int mb = row0 + ((lane >> 4) << 3);
    float bv = (n < Nout) ? bias[n] : 0.f;
    _Float16* p = dst + mb * stride + n;
#pragma unroll
    for (int j = 0; j < 8; ++j) {
        float v = acc[j] + bv;
        v = (v > 0.f) ? v : 0.01f * v;
        p[j * stride] = (_Float16)v;
    }
}

template <int Kpad, int Nout, int padN>
__device__ __forceinline__ void mlp_layer(const _Float16* src, int sstride,
                                          _Float16* dst, int dstride,
                                          const _Float16* Wt, const float* bias,
                                          int lane, int wave) {
#pragma unroll
    for (int c0 = 0; c0 < padN; c0 += 16) {
        v16h bf0 = load_b(Wt, Kpad, Nout, c0, 0, lane);
        v16h bf1 = {};
        if constexpr (Kpad > 32) bf1 = load_b(Wt, Kpad, Nout, c0, 32, lane);
#pragma unroll
        for (int rg = 0; rg < 2; ++rg) {
            int row0 = wave * 32 + rg * 16;
            v8f acc = {};
            v16h a0 = load_a(src, sstride, row0, 0, lane);
            acc = __builtin_amdgcn_wmma_f32_16x16x32_f16(false, a0, false, bf0, (short)0, acc, false, false);
            if constexpr (Kpad > 32) {
                v16h a1 = load_a(src, sstride, row0, 32, lane);
                acc = __builtin_amdgcn_wmma_f32_16x16x32_f16(false, a1, false, bf1, (short)0, acc, false, false);
            }
            store_act(acc, dst, dstride, row0, c0, bias, Nout, lane);
        }
    }
}

__global__ __launch_bounds__(128) void nhgrid_fused(
    const float* __restrict__ x, const float* __restrict__ tables,
    const float* __restrict__ W1, const float* __restrict__ b1,
    const float* __restrict__ W2, const float* __restrict__ b2,
    const float* __restrict__ W3, const float* __restrict__ b3,
    const float* __restrict__ W4, const float* __restrict__ b4,
    const float* __restrict__ W5, const float* __restrict__ b5,
    float* __restrict__ out)
{
    __shared__ _Float16 sFeat[128 * 32];   // encoded features, K = 32
    __shared__ _Float16 sA[128 * 64];      // activation ping
    __shared__ _Float16 sB[128 * 64];      // activation pong
    __shared__ _Float16 sW[4896];          // f16 weights, transposed [Nout][Kpad], K zero-padded
    __shared__ float    sBias[121];

    const int tid  = threadIdx.x;
    const int lane = tid & 31;
    const int wave = tid >> 5;
    const size_t base = (size_t)blockIdx.x * 128;

    // ---- convert weights to f16 LDS, transposed, K padded to mult of 32 ----
    for (int i = tid; i < 64 * 32; i += 128) {           // W1t [64][32]
        int n = i >> 5, k = i & 31; sW[i] = (_Float16)W1[k * 64 + n];
    }
    for (int i = tid; i < 32 * 64; i += 128) {           // W2t [32][64]
        int n = i >> 6, k = i & 63; sW[2048 + i] = (_Float16)W2[k * 32 + n];
    }
    for (int i = tid; i < 16 * 32; i += 128) {           // W3t [16][32]
        int n = i >> 5, k = i & 31; sW[4096 + i] = (_Float16)W3[k * 16 + n];
    }
    for (int i = tid; i < 8 * 32; i += 128) {            // W4t [8][32], K 16 -> pad 32
        int n = i >> 5, k = i & 31; sW[4608 + i] = (_Float16)((k < 16) ? W4[k * 8 + n] : 0.f);
    }
    for (int i = tid; i < 32; i += 128) {                // W5t [1][32], K 8 -> pad 32
        sW[4864 + i] = (_Float16)((i < 8) ? W5[i] : 0.f);
    }
    if (tid < 64) sBias[tid] = b1[tid];
    if (tid < 32) sBias[64 + tid] = b2[tid];
    if (tid < 16) sBias[96 + tid] = b3[tid];
    if (tid < 8)  sBias[112 + tid] = b4[tid];
    if (tid == 0) sBias[120] = b5[0];

    // ---- hash-grid encode: one point per thread ----
    {
        const float SP[16] = {1.f, 1.f, 2.f, 2.f, 3.f, 4.f, 6.f, 7.f,
                              10.f, 13.f, 17.f, 22.f, 29.f, 37.f, 49.f, 64.f};
        const size_t g = base + tid;
        float2 xv = ((const float2*)x)[g];
        float xs0 = xv.x * 0.5f + 0.5f;
        float xs1 = xv.y * 0.5f + 0.5f;
#pragma unroll
        for (int l = 0; l < 16; ++l) {
            float s  = SP[l];
            float fx = xs0 * s, fy = xs1 * s;
            float bxf = floorf(fx), byf = floorf(fy);
            int bx = (int)bxf, by = (int)byf;
            const float2* tb = (const float2*)(tables + ((size_t)l << 20)); // l * H * F
            float  w[4]; float2 tv[4];
            float  wsum = 0.f;
#pragma unroll
            for (int c = 0; c < 4; ++c) {
                int ox = c >> 1, oy = c & 1;                 // corners (0,0),(0,1),(1,0),(1,1)
                float dx = fx - (bxf + (float)ox);
                float dy = fy - (byf + (float)oy);
                float wc = 1.42f - sqrtf(dx * dx + dy * dy); // nonneg (dist <= sqrt2)
                w[c] = wc; wsum += fabsf(wc);
                int hidx = ((bx + ox) + 64 * (by + oy)) & H_MASK;
                tv[c] = tb[hidx];
            }
            float inv = 1.f / fmaxf(wsum, 1e-12f);
            float f0 = 0.f, f1 = 0.f;
#pragma unroll
            for (int c = 0; c < 4; ++c) {
                float wn = w[c] * inv;
                f0 += wn * tv[c].x; f1 += wn * tv[c].y;
            }
            sFeat[tid * 32 + 2 * l]     = (_Float16)f0;
            sFeat[tid * 32 + 2 * l + 1] = (_Float16)f1;
        }
    }
    __syncthreads();  // only cross-wave dep: weights/bias (rows are wave-local)

    // ---- fused MLP, each wave owns rows [wave*32, wave*32+32) ----
    mlp_layer<32, 64, 64>(sFeat, 32, sA, 64, sW,        sBias,       lane, wave); // 32 -> 64
    mlp_layer<64, 32, 32>(sA,    64, sB, 64, sW + 2048, sBias + 64,  lane, wave); // 64 -> 32
    mlp_layer<32, 16, 32>(sB,    64, sA, 64, sW + 4096, sBias + 96,  lane, wave); // 32 -> 16 (pad 32)
    mlp_layer<32,  8, 32>(sA,    64, sB, 64, sW + 4608, sBias + 112, lane, wave); // 16p-> 8  (pad 32)

    // ---- layer 5: 8 (padded 32) -> 1, write f32 to global ----
    {
        v16h bf = load_b(sW + 4864, 32, 1, 0, 0, lane);
        float bv = sBias[120];
#pragma unroll
        for (int rg = 0; rg < 2; ++rg) {
            int row0 = wave * 32 + rg * 16;
            v8f acc = {};
            v16h a = load_a(sB, 64, row0, 0, lane);
            acc = __builtin_amdgcn_wmma_f32_16x16x32_f16(false, a, false, bf, (short)0, acc, false, false);
            if ((lane & 15) == 0) {                       // lanes holding column n == 0
                int mb = row0 + ((lane >> 4) << 3);
#pragma unroll
                for (int j = 0; j < 8; ++j) {
                    float v = acc[j] + bv;
                    v = (v > 0.f) ? v : 0.01f * v;
                    out[base + mb + j] = v;
                }
            }
        }
    }
}

extern "C" void kernel_launch(void* const* d_in, const int* in_sizes, int n_in,
                              void* d_out, int out_size, void* d_ws, size_t ws_size,
                              hipStream_t stream) {
    (void)n_in; (void)d_ws; (void)ws_size; (void)out_size;
    const float* x      = (const float*)d_in[0];
    const float* tables = (const float*)d_in[1];
    const float* W1 = (const float*)d_in[2];  const float* b1 = (const float*)d_in[3];
    const float* W2 = (const float*)d_in[4];  const float* b2 = (const float*)d_in[5];
    const float* W3 = (const float*)d_in[6];  const float* b3 = (const float*)d_in[7];
    const float* W4 = (const float*)d_in[8];  const float* b4 = (const float*)d_in[9];
    const float* W5 = (const float*)d_in[10]; const float* b5 = (const float*)d_in[11];
    float* out = (float*)d_out;

    const int npoints = in_sizes[0] / 2;     // x is [N, 2]
    const int nblocks = npoints / 128;       // 2097152 / 128 = 16384
    nhgrid_fused<<<nblocks, 128, 0, stream>>>(x, tables, W1, b1, W2, b2, W3, b3,
                                              W4, b4, W5, b5, out);
}
